// Attention_60438779789515
// MI455X (gfx1250) — matrix operand order repaired and checked by reading, not compiled, not run on hardware
//
#include <hip/hip_runtime.h>

#define HIDDEN 4096
#define NH 32
#define NKV 8
#define HD 128
#define TSEQ 2048

typedef __attribute__((ext_vector_type(16))) __bf16 bf16x16;
typedef __attribute__((ext_vector_type(8)))  float  f32x8;

union FragAB {
    bf16x16 v;
    uint4   q[2];
};

// Hardware packed f32->bf16 (RTNE): one VALU op for two conversions.
__device__ __forceinline__ unsigned cvt_pk_bf16(float lo, float hi) {
    unsigned r;
    asm("v_cvt_pk_bf16_f32 %0, %1, %2" : "=v"(r) : "v"(lo), "v"(hi));
    return r;
}
__device__ __forceinline__ unsigned short f2bf(float f) {
    return (unsigned short)cvt_pk_bf16(f, f);
}
__device__ __forceinline__ float bf2f(unsigned short h) {
    union { unsigned u; float f; } a; a.u = ((unsigned)h) << 16;
    return a.f;
}

// CDNA5 async vector copy: global -> LDS, 16B per lane, tracked by ASYNCcnt.
__device__ __forceinline__ void async_copy_b128(const void* gaddr, void* lds_ptr) {
    unsigned lds_off = (unsigned)(uintptr_t)lds_ptr;            // low 32 bits = LDS offset
    asm volatile("global_load_async_to_lds_b128 %0, %1, off"
                 : : "v"(lds_off), "v"((unsigned long long)(uintptr_t)gaddr)
                 : "memory");
}
template <int N>
__device__ __forceinline__ void wait_asynccnt() {
    asm volatile("s_wait_asynccnt %0" : : "n"(N) : "memory");
}

// ---------------------------------------------------------------------------
// GEMM-NT: C[M,N] = A[M,K] * B[N,K]^T.  A,B fp32 row-major (K contiguous),
// converted to bf16 (v_cvt_pk_bf16_f32) while staging into LDS.  Global loads
// for slice kk+32 are issued before the WMMA burst of slice kk (register
// pipeline) and kk+64 is prefetched into cache (global_prefetch_b8).
// Block tile 128x256, BK=32, 8 waves; each wave computes 64x64 via 4x4
// wmma f32_16x16x32_bf16 (16 wmma : 16 ds_load_b128).  LDS rows padded to
// 40 halfs (80B) -> conflict-free b128 fragment reads across 64 banks.
// OUT_T stores C transposed (C[n*M+m]); per-lane accumulator rows become
// contiguous so the transposed epilogue is a single b128 store per tile.
// ---------------------------------------------------------------------------
template <bool OUT_BF16, bool OUT_T>
__global__ __launch_bounds__(256) void gemm_nt_kernel(
    const float* __restrict__ A,
    const float* __restrict__ B,
    void* __restrict__ C,
    int M, int N, int K)
{
    constexpr int PA = 40;
    __shared__ __align__(16) unsigned short As[128 * PA];
    __shared__ __align__(16) unsigned short Bs[256 * PA];

    const int tid  = threadIdx.x;
    const int lane = tid & 31;
    const int w    = tid >> 5;
    const int wm   = w & 1;                // 2 wave-rows  x 64
    const int wn   = w >> 1;               // 4 wave-cols  x 64
    const int m0 = blockIdx.y * 128;
    const int n0 = blockIdx.x * 256;

    f32x8 acc[4][4];
#pragma unroll
    for (int i = 0; i < 4; i++)
#pragma unroll
        for (int j = 0; j < 4; j++)
#pragma unroll
            for (int e = 0; e < 8; e++) acc[i][j][e] = 0.0f;

    // staging map: A row tid/2 (16 floats at 16*(tid&1)); B row tid (32 floats)
    const int ar = tid >> 1, ac = (tid & 1) * 16;
    const int br = tid;

    float4 aR[4], bR[8];
    {
        const float4* sa = (const float4*)(A + (size_t)(m0 + ar) * K + ac);
        const float4* sb = (const float4*)(B + (size_t)(n0 + br) * K);
#pragma unroll
        for (int i = 0; i < 4; i++) aR[i] = sa[i];
#pragma unroll
        for (int i = 0; i < 8; i++) bR[i] = sb[i];
    }

    const int arow = lane & 15;
    const int akb  = (lane < 16) ? 0 : 8;
    const int bkb  = (lane < 16) ? 0 : 8;

    for (int kk = 0; kk < K; kk += 32) {
        {
            unsigned t[8];
            const float* f = (const float*)aR;
#pragma unroll
            for (int i = 0; i < 8; i++) t[i] = cvt_pk_bf16(f[2 * i], f[2 * i + 1]);
            uint4* d = (uint4*)&As[ar * PA + ac];
            d[0] = *(const uint4*)&t[0];
            d[1] = *(const uint4*)&t[4];
        }
        {
            unsigned t[16];
            const float* f = (const float*)bR;
#pragma unroll
            for (int i = 0; i < 16; i++) t[i] = cvt_pk_bf16(f[2 * i], f[2 * i + 1]);
            uint4* d = (uint4*)&Bs[br * PA];
#pragma unroll
            for (int i = 0; i < 4; i++) d[i] = *(const uint4*)&t[4 * i];
        }
        __syncthreads();

        if (kk + 32 < K) {   // register pipeline: fetch next slice now
            const float4* sa = (const float4*)(A + (size_t)(m0 + ar) * K + kk + 32 + ac);
            const float4* sb = (const float4*)(B + (size_t)(n0 + br) * K + kk + 32);
#pragma unroll
            for (int i = 0; i < 4; i++) aR[i] = sa[i];
#pragma unroll
            for (int i = 0; i < 8; i++) bR[i] = sb[i];
        }
        if (kk + 64 < K) {   // cache prefetch two slices ahead
            __builtin_prefetch(A + (size_t)(m0 + ar) * K + kk + 64 + ac, 0, 0);
            __builtin_prefetch(B + (size_t)(n0 + br) * K + kk + 64, 0, 0);
        }

        FragAB af[4];
#pragma unroll
        for (int mt = 0; mt < 4; mt++) {
            const int r = wm * 64 + mt * 16 + arow;
            af[mt].q[0] = *(const uint4*)&As[r * PA + akb];
            af[mt].q[1] = *(const uint4*)&As[r * PA + akb + 16];
        }
#pragma unroll
        for (int nt = 0; nt < 4; nt++) {
            FragAB bf;
            const int c = wn * 64 + nt * 16 + arow;
            bf.q[0] = *(const uint4*)&Bs[c * PA + bkb];
            bf.q[1] = *(const uint4*)&Bs[c * PA + bkb + 16];
#pragma unroll
            for (int mt = 0; mt < 4; mt++) {
                acc[mt][nt] = __builtin_amdgcn_wmma_f32_16x16x32_bf16(
                    false, af[mt].v, false, bf.v,
                    (short)0, acc[mt][nt], false, false);
            }
        }
        __syncthreads();
    }

    const int rh = (lane >> 4) << 3;
#pragma unroll
    for (int mt = 0; mt < 4; mt++)
#pragma unroll
        for (int nt = 0; nt < 4; nt++) {
            const int col = n0 + wn * 64 + nt * 16 + (lane & 15);
            const int rowb = m0 + wm * 64 + mt * 16 + rh;
            if (OUT_T) {                    // contiguous along rows -> b128 store
                unsigned t4[4];
#pragma unroll
                for (int v = 0; v < 4; v++)
                    t4[v] = cvt_pk_bf16(acc[mt][nt][2 * v], acc[mt][nt][2 * v + 1]);
                *(uint4*)&((unsigned short*)C)[(size_t)col * M + rowb] =
                    *(const uint4*)t4;
            } else {
#pragma unroll
                for (int v = 0; v < 8; v++) {
                    const float val = acc[mt][nt][v];
                    if (OUT_BF16)
                        ((unsigned short*)C)[(size_t)(rowb + v) * N + col] = f2bf(val);
                    else
                        ((float*)C)[(size_t)(rowb + v) * N + col] = val;
                }
            }
        }
}

// ---------------------------------------------------------------------------
// RoPE, in-place on bf16 [T, nh, 128]; one thread per rotation pair (j, j+64).
// inv_freq computed in base-2 domain (hardware exp is exp2).
// ---------------------------------------------------------------------------
__global__ void rope_kernel(unsigned short* __restrict__ buf, int T, int nh)
{
    const int idx = blockIdx.x * blockDim.x + threadIdx.x;
    const int ppt = nh * 64;
    const int t = idx / ppt;
    const int r = idx - t * ppt;
    const int h = r >> 6;
    const int j = r & 63;
    const float LOG2_BASE_OVER_HALFD = 0.20762051f;   // log2(10000)/64
    const float ang = (float)t * __builtin_amdgcn_exp2f(-(float)j * LOG2_BASE_OVER_HALFD);
    float s, c;
    __sincosf(ang, &s, &c);
    const size_t base = (size_t)t * nh * 128 + h * 128 + j;
    const float x1 = bf2f(buf[base]);
    const float x2 = bf2f(buf[base + 64]);
    buf[base]      = f2bf(x1 * c - x2 * s);
    buf[base + 64] = f2bf(x1 * s + x2 * c);
}

// ---------------------------------------------------------------------------
// Flash attention (causal, GQA).  Block = 4 waves = 64 query rows of one head.
// Q fragments resident in VGPRs.  K tile double-buffered in LDS via
// global_load_async_to_lds_b128 (next tile prefetched during compute); V tile
// (pre-transposed in global: Vt[feature][t]) prefetched during the S/softmax
// phase of the same iteration.  V ops are issued before next-K ops so the
// in-order ASYNCcnt retirement allows s_wait_asynccnt<=8 to release V while
// next-K is still in flight.  Softmax runs in the exp2 domain (log2e folded
// into the 1/sqrt(d) scale).
// ---------------------------------------------------------------------------
__global__ __launch_bounds__(128) void flash_kernel(
    const unsigned short* __restrict__ Q,    // [T, 32*128] bf16 (post-RoPE)
    const unsigned short* __restrict__ Kb,   // [T,  8*128] bf16 (post-RoPE)
    const unsigned short* __restrict__ Vt,   // [8*128, T]  bf16 (pre-transposed)
    float* __restrict__ O)                   // [T, 32*128] f32
{
    constexpr int PK = 136;   // 64 key rows x 128 halfs (272B pitch)
    constexpr int PV = 72;    // 128 d rows  x  64 halfs (144B pitch)
    constexpr int PS = 72;    // 64 P rows   x  64 halfs
    __shared__ __align__(16) unsigned short Ks[2][64 * PK];
    __shared__ __align__(16) unsigned short Vs[128 * PV];
    __shared__ __align__(16) unsigned short Ss[64 * PS];

    const int qt   = blockIdx.x;
    const int h    = blockIdx.y;
    const int kvh  = h >> 2;
    const int tid  = threadIdx.x;
    const int lane = tid & 31;
    const int w    = tid >> 5;
    const int half8 = (lane >> 4) << 3;

    const float scale2 = 0.12751740f;   // (1/sqrt(128)) * log2(e)

    // ---- async staging helpers (8 x b128 per thread each) ----
    const int kr = tid >> 1, kseg = (tid & 1) * 64;   // K: row, 64-half segment
    auto stage_K = [&](int j, int buf) {
        const unsigned short* src = Kb + (size_t)(j * 64 + kr) * (NKV * HD)
                                    + kvh * HD + kseg;
#pragma unroll
        for (int i = 0; i < 8; i++)
            async_copy_b128(src + i * 8, &Ks[buf][kr * PK + kseg + i * 8]);
    };
    auto stage_V = [&](int j) {
        const unsigned short* src = Vt + (size_t)(kvh * HD + tid) * TSEQ + j * 64;
#pragma unroll
        for (int i = 0; i < 8; i++)
            async_copy_b128(src + i * 8, &Vs[tid * PV + i * 8]);
    };

    // resident Q fragments: 16 rows x 128 -> 4 fragments of 16x32
    FragAB qa[4];
    {
        const int row = qt * 64 + w * 16 + (lane & 15);
        const unsigned short* qrow = Q + (size_t)row * (NH * HD) + h * HD;
        const int kb = (lane < 16) ? 0 : 8;
#pragma unroll
        for (int kf = 0; kf < 4; kf++) {
            qa[kf].q[0] = *(const uint4*)&qrow[kf * 32 + kb];
            qa[kf].q[1] = *(const uint4*)&qrow[kf * 32 + kb + 16];
        }
    }

    f32x8 oacc[8];
#pragma unroll
    for (int i = 0; i < 8; i++)
#pragma unroll
        for (int e = 0; e < 8; e++) oacc[i][e] = 0.0f;
    float mrow[8], lrow[8];
#pragma unroll
    for (int v = 0; v < 8; v++) { mrow[v] = -3.0e38f; lrow[v] = 0.0f; }

    stage_K(0, 0);

    for (int j = 0; j <= qt; j++) {
        const int buf = j & 1;
        wait_asynccnt<0>();       // K(j) resident everywhere in this thread
        __syncthreads();          // all threads staged; prev-iter LDS reads done

        stage_V(j);               // async: V needed only in the PV phase
        if (j < qt) stage_K(j + 1, buf ^ 1);

        // ---- S = Q * K^T (16x64 per wave) ----
        f32x8 sacc[4];
#pragma unroll
        for (int nt = 0; nt < 4; nt++)
#pragma unroll
            for (int e = 0; e < 8; e++) sacc[nt][e] = 0.0f;
#pragma unroll
        for (int kf = 0; kf < 4; kf++) {
#pragma unroll
            for (int nt = 0; nt < 4; nt++) {
                FragAB bfr;
                const int c  = nt * 16 + (lane & 15);
                const int k0 = kf * 32 + ((lane < 16) ? 0 : 8);
                bfr.q[0] = *(const uint4*)&Ks[buf][c * PK + k0];
                bfr.q[1] = *(const uint4*)&Ks[buf][c * PK + k0 + 16];
                sacc[nt] = __builtin_amdgcn_wmma_f32_16x16x32_bf16(
                    false, qa[kf].v, false, bfr.v,
                    (short)0, sacc[nt], false, false);
            }
        }

        // ---- scale (exp2 domain) + causal mask (diagonal block only) ----
        const int myrow = qt * 64 + w * 16 + half8;
#pragma unroll
        for (int nt = 0; nt < 4; nt++) {
            const int col = j * 64 + nt * 16 + (lane & 15);
#pragma unroll
            for (int v = 0; v < 8; v++) {
                float x = sacc[nt][v] * scale2;
                if (j == qt && col > myrow + v) x = -3.0e38f;
                sacc[nt][v] = x;
            }
        }

        // ---- online softmax (row stats replicated per 16-lane half) ----
        float alpha[8];
#pragma unroll
        for (int v = 0; v < 8; v++) {
            float x = fmaxf(fmaxf(sacc[0][v], sacc[1][v]),
                            fmaxf(sacc[2][v], sacc[3][v]));
#pragma unroll
            for (int msk = 1; msk < 16; msk <<= 1)
                x = fmaxf(x, __shfl_xor(x, msk, 32));
            const float mnew = fmaxf(mrow[v], x);
            alpha[v] = __builtin_amdgcn_exp2f(mrow[v] - mnew);
            mrow[v]  = mnew;
            lrow[v] *= alpha[v];
        }
#pragma unroll
        for (int nt = 0; nt < 4; nt++)
#pragma unroll
            for (int v = 0; v < 8; v++)
                sacc[nt][v] = __builtin_amdgcn_exp2f(sacc[nt][v] - mrow[v]);
#pragma unroll
        for (int v = 0; v < 8; v++) {
            float rs = sacc[0][v] + sacc[1][v] + sacc[2][v] + sacc[3][v];
#pragma unroll
            for (int msk = 1; msk < 16; msk <<= 1)
                rs += __shfl_xor(rs, msk, 32);
            lrow[v] += rs;
        }
#pragma unroll
        for (int d = 0; d < 8; d++)
#pragma unroll
            for (int v = 0; v < 8; v++) oacc[d][v] *= alpha[v];

        // ---- P (C layout) -> LDS -> A layout ----
#pragma unroll
        for (int nt = 0; nt < 4; nt++)
#pragma unroll
            for (int v = 0; v < 8; v++) {
                const int rr = w * 16 + v + half8;
                const int cc = nt * 16 + (lane & 15);
                Ss[rr * PS + cc] = f2bf(sacc[nt][v]);
            }

        if (j < qt) wait_asynccnt<8>();   // drain V(j); next-K may still fly
        else        wait_asynccnt<0>();
        __syncthreads();                  // V staged block-wide; Ss ordered

        // ---- O += P * V ----
#pragma unroll
        for (int kf = 0; kf < 2; kf++) {
            FragAB pa;
            const int rr = w * 16 + (lane & 15);
            const int k0 = kf * 32 + ((lane < 16) ? 0 : 8);
            pa.q[0] = *(const uint4*)&Ss[rr * PS + k0];
            pa.q[1] = *(const uint4*)&Ss[rr * PS + k0 + 16];
#pragma unroll
            for (int nt = 0; nt < 8; nt++) {
                FragAB vb;
                const int dcol = nt * 16 + (lane & 15);
                const int kk0  = kf * 32 + ((lane < 16) ? 0 : 8);
                vb.q[0] = *(const uint4*)&Vs[dcol * PV + kk0];
                vb.q[1] = *(const uint4*)&Vs[dcol * PV + kk0 + 16];
                oacc[nt] = __builtin_amdgcn_wmma_f32_16x16x32_bf16(
                    false, pa.v, false, vb.v,
                    (short)0, oacc[nt], false, false);
            }
        }
    }

    // ---- normalize and write f32 attention output [T, 32*128] ----
#pragma unroll
    for (int nt = 0; nt < 8; nt++)
#pragma unroll
        for (int v = 0; v < 8; v++) {
            const int row = qt * 64 + w * 16 + v + half8;
            const int col = h * HD + nt * 16 + (lane & 15);
            O[(size_t)row * (NH * HD) + col] = oacc[nt][v] / lrow[v];
        }
}

// ---------------------------------------------------------------------------
extern "C" void kernel_launch(void* const* d_in, const int* in_sizes, int n_in,
                              void* d_out, int out_size, void* d_ws, size_t ws_size,
                              hipStream_t stream)
{
    (void)in_sizes; (void)n_in; (void)out_size; (void)ws_size;
    const float* hidden = (const float*)d_in[0];
    const float* Wq = (const float*)d_in[1];
    const float* Wk = (const float*)d_in[2];
    const float* Wv = (const float*)d_in[3];
    const float* Wo = (const float*)d_in[4];
    float* out = (float*)d_out;

    unsigned short* Qb  = (unsigned short*)d_ws;                   // [T, 4096] bf16
    unsigned short* Kbf = Qb  + (size_t)TSEQ * NH  * HD;           // [T, 1024] bf16
    unsigned short* Vtg = Kbf + (size_t)TSEQ * NKV * HD;           // [1024, T] bf16 (transposed)
    float*          att = (float*)(Vtg + (size_t)TSEQ * NKV * HD); // [T, 4096] f32

    dim3 blk(256);
    gemm_nt_kernel<true, false><<<dim3(NH * HD / 256, TSEQ / 128), blk, 0, stream>>>(
        hidden, Wq, Qb, TSEQ, NH * HD, HIDDEN);
    gemm_nt_kernel<true, false><<<dim3(NKV * HD / 256, TSEQ / 128), blk, 0, stream>>>(
        hidden, Wk, Kbf, TSEQ, NKV * HD, HIDDEN);
    gemm_nt_kernel<true, true><<<dim3(NKV * HD / 256, TSEQ / 128), blk, 0, stream>>>(
        hidden, Wv, Vtg, TSEQ, NKV * HD, HIDDEN);

    rope_kernel<<<(TSEQ * NH  * 64) / 256, 256, 0, stream>>>(Qb,  TSEQ, NH);
    rope_kernel<<<(TSEQ * NKV * 64) / 256, 256, 0, stream>>>(Kbf, TSEQ, NKV);

    flash_kernel<<<dim3(TSEQ / 64, NH), 128, 0, stream>>>(Qb, Kbf, Vtg, att);

    gemm_nt_kernel<false, false><<<dim3(HIDDEN / 256, TSEQ / 128), blk, 0, stream>>>(
        att, Wo, out, TSEQ, HIDDEN, HIDDEN);
}
